// CrystalGraphConvNet_57604101374550
// MI455X (gfx1250) — matrix operand order repaired and checked by reading, not compiled, not run on hardware
//
#include <hip/hip_runtime.h>
#include <hip/hip_bf16.h>

// ---------------------------------------------------------------------------
// CGCNN forward for MI455X (gfx1250): wave32, WMMA bf16 (f32 accum),
// async global->LDS double-buffered staging for the conv gather/stream.
// ---------------------------------------------------------------------------

typedef __attribute__((ext_vector_type(16))) __bf16 v16bf;
typedef __attribute__((ext_vector_type(8)))  float  v8f;

#define NATOM   100000
#define NBR_M   12
#define FEAT    64
#define BOND    41
#define CIN     169      // 2F + B
#define KPAD    192      // CIN padded to multiple of 32
#define NCRYS   2000
#define EMB_IN  92
#define EMB_KP  96
#define A_STRIDE   200   // bf16 elems per LDS row (>=KPAD, keeps 16B alignment)
#define G_STRIDE   132   // f32 elems per LDS gact row

// staging layout (floats): [xself 4x64 | xnbr 48x64 | bond 48x41]
#define SX      0
#define XN      256
#define BD      3328     // 256 + 48*64
#define STAGE_F 5296     // 3328 + 48*41

#define NB128   832      // 64 (xself) + 768 (xnbr) 16-byte transfers
#define NI128   4        // ceil(832/256)
#define NB32    1968     // 48*41 bond words
#define NI32    8        // ceil(1968/256)
// batch per wave = NI128 + NI32 = 12 async instructions

// round-to-nearest-even float -> bf16
__device__ __forceinline__ __bf16 f2bf(float f) {
    union { float f; unsigned u; } v; v.f = f;
    unsigned r = (v.u + 0x7FFFu + ((v.u >> 16) & 1u)) >> 16;
    unsigned short h = (unsigned short)r;
    return __builtin_bit_cast(__bf16, h);
}
__device__ __forceinline__ float sigmoidf_(float x) {
    return 1.0f / (1.0f + __expf(-x));
}
__device__ __forceinline__ float softplusf_(float x) {
    return fmaxf(x, 0.0f) + log1pf(__expf(-fabsf(x)));
}

// CDNA5 async global->LDS (ASYNCcnt-tracked), per-lane LDS + global addresses.
__device__ __forceinline__ void async_b128(uint32_t lds, uint64_t gaddr) {
    asm volatile("global_load_async_to_lds_b128 %0, %1, off"
                 :: "v"(lds), "v"(gaddr) : "memory");
}
__device__ __forceinline__ void async_b32(uint32_t lds, uint64_t gaddr) {
    asm volatile("global_load_async_to_lds_b32 %0, %1, off"
                 :: "v"(lds), "v"(gaddr) : "memory");
}

// ---------------------------------------------------------------------------
// Weight packing
// ---------------------------------------------------------------------------
__global__ void pack_conv_w(const float* __restrict__ w, __bf16* __restrict__ out) {
    int i = blockIdx.x * blockDim.x + threadIdx.x;
    if (i >= 3 * KPAD * 128) return;
    int c = i % 128, k = (i / 128) % KPAD, l = i / (128 * KPAD);
    float v = (k < CIN) ? w[(l * CIN + k) * 128 + c] : 0.0f;
    out[i] = f2bf(v);
}
__global__ void pack_emb_w(const float* __restrict__ w, __bf16* __restrict__ out) {
    int i = blockIdx.x * blockDim.x + threadIdx.x;
    if (i >= EMB_KP * 64) return;
    int f = i % 64, k = i / 64;
    out[i] = f2bf((k < EMB_IN) ? w[k * 64 + f] : 0.0f);
}

// ---------------------------------------------------------------------------
// Embedding: x = atom_fea(100000x92) @ emb_w(92x64) + emb_b (WMMA bf16),
// coalesced b128 stores via LDS bounce.
// ---------------------------------------------------------------------------
__global__ __launch_bounds__(256)
void embed_kernel(const float* __restrict__ atom_fea,
                  const __bf16* __restrict__ Ebf,   // [96][64]
                  const float* __restrict__ emb_b,
                  float* __restrict__ x) {
    __shared__ __align__(16) __bf16 lA[32 * 104];
    __shared__ __align__(16) float xout[32 * 68];
    const int tid = threadIdx.x, wave = tid >> 5, lane = tid & 31;
    const int ct = wave & 3, rt = wave >> 2;
    const int col0 = ct * 16;
    const int n16 = lane & 15, half = lane >> 4;

    v16bf Bfrag[3];
    for (int c = 0; c < 3; ++c) {
        const int kb = c * 32 + half * 16;
        v16bf b;
#pragma unroll
        for (int i = 0; i < 16; ++i) b[i] = Ebf[(kb + i) * 64 + col0 + n16];
        Bfrag[c] = b;
    }
    const int mycol = col0 + n16;
    const float bias = emb_b[mycol];

    const int ntiles = NATOM / 32;
    for (int tile = blockIdx.x; tile < ntiles; tile += gridDim.x) {
        __syncthreads();
        for (int r = wave; r < 32; r += 8) {
            const float* src = atom_fea + (size_t)(tile * 32 + r) * EMB_IN;
            __bf16* dst = &lA[r * 104];
            dst[lane]      = f2bf(src[lane]);
            dst[32 + lane] = f2bf(src[32 + lane]);
            if (lane < 28) dst[64 + lane] = f2bf(src[64 + lane]);
            if (lane < 4)  dst[92 + lane] = f2bf(0.0f);
        }
        __syncthreads();
        // hoist all A-fragment LDS loads, then run the dependent WMMA chain
        v16bf Afrag[3];
#pragma unroll
        for (int c = 0; c < 3; ++c) {
            const __bf16* rp = &lA[(rt * 16 + n16) * 104 + c * 32 + half * 16];
#pragma unroll
            for (int i = 0; i < 8; ++i) { Afrag[c][i] = rp[i]; Afrag[c][8 + i] = rp[8 + i]; }
        }
        v8f Cacc = {};
#pragma unroll
        for (int c = 0; c < 3; ++c)
            Cacc = __builtin_amdgcn_wmma_f32_16x16x32_bf16(
                false, Afrag[c], false, Bfrag[c], (short)0, Cacc, false, false);
#pragma unroll
        for (int v = 0; v < 8; ++v) {
            const int row = rt * 16 + half * 8 + v;
            xout[row * 68 + mycol] = Cacc[v] + bias;
        }
        __syncthreads();
        // coalesced stores: thread t -> row t>>3, 8 cols (two b128)
        {
            const int row = tid >> 3, c0 = (tid & 7) * 8;
            const float4* s4 = (const float4*)&xout[row * 68 + c0];
            float4* d4 = (float4*)&x[(size_t)(tile * 32 + row) * FEAT + c0];
            d4[0] = s4[0];
            d4[1] = s4[1];
        }
    }
}

// ---------------------------------------------------------------------------
// Conv GEMM passes. Quad = 4 atoms = 48 rows of A = [x[n] | x[nbr] | nbr_fea].
// Raw f32 payload is async-copied global->LDS (double buffered), then
// converted to the WMMA bf16 A-tile in LDS.
// PASS2=false: per-channel sum/sumsq of g. PASS2=true: BN + gated sum over M.
// ---------------------------------------------------------------------------
template <bool PASS2>
__global__ __launch_bounds__(256)
void conv_pass(const float* __restrict__ x,
               const float* __restrict__ nbr_fea,
               const int*   __restrict__ nbr_idx,
               const __bf16* __restrict__ Wbf,    // [KPAD][128]
               const float* __restrict__ conv_b,  // [128]
               float* __restrict__ stats,         // [2*128] (pass1)
               const float* __restrict__ bn_s,    // [128]   (pass2)
               const float* __restrict__ bn_t,    // [128]   (pass2)
               float* __restrict__ nbr_sumed) {   // [N][64] (pass2)
    __shared__ __align__(16) float  stage[2][STAGE_F];
    __shared__ __align__(16) __bf16 lA[48 * A_STRIDE];
    __shared__ __align__(16) float  gact[48 * G_STRIDE];
    __shared__ __align__(16) float  dummy[8];

    const int tid = threadIdx.x, wave = tid >> 5, lane = tid & 31;
    const int col0 = wave * 16;
    const int n16 = lane & 15, half = lane >> 4;

    v16bf Bfrag[6];
    for (int c = 0; c < 6; ++c) {
        const int kb = c * 32 + half * 16;
        v16bf b;
#pragma unroll
        for (int i = 0; i < 16; ++i) b[i] = Wbf[(kb + i) * 128 + col0 + n16];
        Bfrag[c] = b;
    }
    const int mycol = col0 + n16;
    const float bias = conv_b[mycol];
    float s = 0.f, t = 0.f;
    if (PASS2) { s = bn_s[mycol]; t = bn_t[mycol]; }
    float accS = 0.f, accQ = 0.f;

    // zero-pad K in [169,192) once; never rewritten
    for (int r = wave; r < 48; r += 8)
        if (lane < 23) lA[r * A_STRIDE + 169 + lane] = f2bf(0.0f);

    const uint32_t dummy_lds = (uint32_t)(uintptr_t)&dummy[0];
    const uint64_t xbase = (uint64_t)(uintptr_t)x;
    const uint64_t bbase = (uint64_t)(uintptr_t)nbr_fea;

    auto issue_quad = [&](int qn, int b) {
        const uint32_t sbase = (uint32_t)(uintptr_t)&stage[b][0];
        // b128 part: xself (64) + gathered xnbr (768)
#pragma unroll
        for (int k = 0; k < NI128; ++k) {
            const int idx = k * 256 + tid;
            uint32_t lds = dummy_lds;
            uint64_t ga  = xbase;
            if (idx < 64) {
                lds = sbase + (uint32_t)idx * 16u;
                ga  = xbase + (uint64_t)qn * 1024u + (uint32_t)idx * 16u;
            } else if (idx < NB128) {
                const int i = idx - 64;
                const int row = i >> 4, c = i & 15;
                const int j = nbr_idx[qn * 48 + row];
                lds = sbase + (uint32_t)(XN * 4) + (uint32_t)row * 256u + (uint32_t)c * 16u;
                ga  = xbase + (uint64_t)(uint32_t)j * 256u + (uint32_t)c * 16u;
            }
            async_b128(lds, ga);
        }
        // b32 part: bond features (48 rows x 41 words; rows only 4B aligned)
#pragma unroll
        for (int k = 0; k < NI32; ++k) {
            const int idx = k * 256 + tid;
            uint32_t lds = dummy_lds;
            uint64_t ga  = xbase;
            if (idx < NB32) {
                const int row = idx / BOND, c = idx - row * BOND;
                lds = sbase + (uint32_t)(BD * 4) + (uint32_t)(row * BOND + c) * 4u;
                ga  = bbase + (uint64_t)(qn * 48 + row) * 164u + (uint32_t)c * 4u;
            }
            async_b32(lds, ga);
        }
    };

    const int nquads = NATOM / 4;
    const int stride = gridDim.x;
    if ((int)blockIdx.x < nquads) issue_quad(blockIdx.x, 0);
    int buf = 0;
    for (int q = blockIdx.x; q < nquads; q += stride) {
        const int qn = q + stride;
        if (qn < nquads) {
            issue_quad(qn, buf ^ 1);
            asm volatile("s_wait_asynccnt 0xc" ::: "memory"); // prev batch (12) drained
        } else {
            asm volatile("s_wait_asynccnt 0x0" ::: "memory");
        }
        __syncthreads();  // all waves' async batches for `buf` visible

        // convert staged f32 -> bf16 A tile (wave per row)
        const float* st = &stage[buf][0];
        for (int r = wave; r < 48; r += 8) {
            __bf16* dst = &lA[r * A_STRIDE];
            const int a64 = (r / 12) * 64;
            dst[lane]       = f2bf(st[SX + a64 + lane]);
            dst[32 + lane]  = f2bf(st[SX + a64 + 32 + lane]);
            dst[64 + lane]  = f2bf(st[XN + r * 64 + lane]);
            dst[96 + lane]  = f2bf(st[XN + r * 64 + 32 + lane]);
            dst[128 + lane] = f2bf(st[BD + r * BOND + lane]);
            if (lane < 9) dst[160 + lane] = f2bf(st[BD + r * BOND + 32 + lane]);
        }
        __syncthreads();

        for (int rt = 0; rt < 3; ++rt) {
            // hoist all 12 ds_load_b128 for this row-tile, then 6 WMMAs back-to-back
            v16bf Afrag[6];
#pragma unroll
            for (int c = 0; c < 6; ++c) {
                const __bf16* rp = &lA[(rt * 16 + n16) * A_STRIDE + c * 32 + half * 16];
#pragma unroll
                for (int i = 0; i < 8; ++i) { Afrag[c][i] = rp[i]; Afrag[c][8 + i] = rp[8 + i]; }
            }
            v8f Cacc = {};
#pragma unroll
            for (int c = 0; c < 6; ++c)
                Cacc = __builtin_amdgcn_wmma_f32_16x16x32_bf16(
                    false, Afrag[c], false, Bfrag[c], (short)0, Cacc, false, false);
            if (!PASS2) {
#pragma unroll
                for (int v = 0; v < 8; ++v) {
                    float g = Cacc[v] + bias;
                    accS += g; accQ += g * g;
                }
            } else {
#pragma unroll
                for (int v = 0; v < 8; ++v) {
                    float g = (Cacc[v] + bias) * s + t;
                    float act = (col0 < 64) ? sigmoidf_(g) : softplusf_(g);
                    const int row = rt * 16 + half * 8 + v;
                    gact[row * G_STRIDE + mycol] = act;
                }
            }
        }
        if (PASS2) {
            __syncthreads();
            const int a = tid >> 6, f = tid & 63;
            float sum = 0.f;
#pragma unroll
            for (int m = 0; m < NBR_M; ++m) {
                const int row = a * NBR_M + m;
                sum += gact[row * G_STRIDE + f] * gact[row * G_STRIDE + 64 + f];
            }
            nbr_sumed[(size_t)(q * 4 + a) * FEAT + f] = sum;
        }
        buf ^= 1;
    }
    if (!PASS2) {
        atomicAdd(&stats[mycol], accS);
        atomicAdd(&stats[128 + mycol], accQ);
    }
}

// ---------------------------------------------------------------------------
// BN helpers
// ---------------------------------------------------------------------------
__global__ void finalize_bn(const float* __restrict__ stats,
                            const float* __restrict__ gamma,
                            const float* __restrict__ beta,
                            float cnt_inv, int nch,
                            float* __restrict__ s_out, float* __restrict__ t_out) {
    int c = threadIdx.x;
    if (c < nch) {
        float mean = stats[c] * cnt_inv;
        float var  = stats[nch + c] * cnt_inv - mean * mean;
        float sc   = gamma[c] * rsqrtf(var + 1e-5f);
        s_out[c] = sc;
        t_out[c] = beta[c] - mean * sc;
    }
}

__global__ __launch_bounds__(256)
void col_stats64(const float* __restrict__ a, int nrows, float* __restrict__ stats) {
    int tid = blockIdx.x * blockDim.x + threadIdx.x;
    int f = tid & 63;
    int r0 = tid >> 6;
    int stride = (gridDim.x * blockDim.x) >> 6;
    float s = 0.f, q = 0.f;
    for (int r = r0; r < nrows; r += stride) {
        float v = a[(size_t)r * 64 + f];
        s += v; q += v * v;
    }
    atomicAdd(&stats[f], s);
    atomicAdd(&stats[64 + f], q);
}

__global__ void x_update(float* __restrict__ x, const float* __restrict__ ns,
                         const float* __restrict__ s, const float* __restrict__ t) {
    int i = blockIdx.x * blockDim.x + threadIdx.x;
    if (i >= NATOM * FEAT) return;
    int f = i & 63;
    float v = x[i] + ns[i] * s[f] + t[f];
    x[i] = softplusf_(v);
}

// ---------------------------------------------------------------------------
// Segment mean pool + 3-layer head
// ---------------------------------------------------------------------------
__global__ void pool_kernel(const float* __restrict__ x, const int* __restrict__ seg,
                            float* __restrict__ sums, float* __restrict__ cnts) {
    int i = blockIdx.x * blockDim.x + threadIdx.x;
    if (i >= NATOM * FEAT) return;
    int n = i >> 6, f = i & 63;
    int c = seg[n];
    atomicAdd(&sums[(size_t)c * FEAT + f], x[i]);
    if (f == 0) atomicAdd(&cnts[c], 1.0f);
}

__global__ __launch_bounds__(256)
void head_kernel(const float* __restrict__ sums, const float* __restrict__ cnts,
                 const float* __restrict__ fc_w, const float* __restrict__ fc_b,
                 const float* __restrict__ head_w, const float* __restrict__ head_b,
                 const float* __restrict__ out_w, const float* __restrict__ out_b,
                 float* __restrict__ out) {
    __shared__ float crys[64];
    __shared__ float h1[128];
    __shared__ float red[256];
    const int c = blockIdx.x, t = threadIdx.x;
    float inv = 1.0f / fmaxf(cnts[c], 1.0f);
    if (t < 64) crys[t] = sums[(size_t)c * FEAT + t] * inv;
    __syncthreads();
    if (t < 128) {
        float a = fc_b[t];
        for (int k = 0; k < 64; ++k) a += crys[k] * fc_w[k * 128 + t];
        h1[t] = fmaxf(a, 0.0f);
    }
    __syncthreads();
    float h2 = head_b[t];
    for (int k = 0; k < 128; ++k) h2 += h1[k] * head_w[k * 256 + t];
    red[t] = h2 * out_w[t];
    __syncthreads();
    for (int off = 128; off > 0; off >>= 1) {
        if (t < off) red[t] += red[t + off];
        __syncthreads();
    }
    if (t == 0) out[c] = red[0] + out_b[0];
}

// ---------------------------------------------------------------------------
// Orchestration
// ---------------------------------------------------------------------------
extern "C" void kernel_launch(void* const* d_in, const int* in_sizes, int n_in,
                              void* d_out, int out_size, void* d_ws, size_t ws_size,
                              hipStream_t stream) {
    const float* atom_fea = (const float*)d_in[0];
    const float* nbr_fea  = (const float*)d_in[1];
    const int*   nbr_idx  = (const int*)d_in[2];
    const int*   cry_idx  = (const int*)d_in[3];
    const float* emb_w    = (const float*)d_in[4];
    const float* emb_b    = (const float*)d_in[5];
    const float* conv_w   = (const float*)d_in[6];
    const float* conv_b   = (const float*)d_in[7];
    const float* bn1_g    = (const float*)d_in[8];
    const float* bn1_b    = (const float*)d_in[9];
    const float* bn2_g    = (const float*)d_in[10];
    const float* bn2_b    = (const float*)d_in[11];
    const float* fc_w     = (const float*)d_in[12];
    const float* fc_b     = (const float*)d_in[13];
    const float* head_w   = (const float*)d_in[14];
    const float* head_b   = (const float*)d_in[15];
    const float* out_w    = (const float*)d_in[16];
    const float* out_b    = (const float*)d_in[17];
    float* out = (float*)d_out;

    char* p = (char*)d_ws;
    float* x      = (float*)p; p += (size_t)NATOM * FEAT * 4;
    float* ns     = (float*)p; p += (size_t)NATOM * FEAT * 4;
    float* sums   = (float*)p; p += (size_t)NCRYS * FEAT * 4;
    float* cnts   = (float*)p; p += (size_t)NCRYS * 4;
    float* stats1 = (float*)p; p += 256 * 4;
    float* stats2 = (float*)p; p += 128 * 4;
    float* s1     = (float*)p; p += 128 * 4;
    float* t1     = (float*)p; p += 128 * 4;
    float* s2     = (float*)p; p += 64 * 4;
    float* t2     = (float*)p; p += 64 * 4;
    __bf16* Wbf   = (__bf16*)p; p += (size_t)3 * KPAD * 128 * 2;
    __bf16* Ebf   = (__bf16*)p; p += (size_t)EMB_KP * 64 * 2;

    pack_conv_w<<<(3 * KPAD * 128 + 255) / 256, 256, 0, stream>>>(conv_w, Wbf);
    pack_emb_w<<<(EMB_KP * 64 + 255) / 256, 256, 0, stream>>>(emb_w, Ebf);

    embed_kernel<<<1024, 256, 0, stream>>>(atom_fea, Ebf, emb_b, x);

    for (int l = 0; l < 3; ++l) {
        const __bf16* Wl = Wbf + (size_t)l * KPAD * 128;
        const float*  bl = conv_b + l * 128;
        hipMemsetAsync(stats1, 0, 256 * 4, stream);
        conv_pass<false><<<1024, 256, 0, stream>>>(
            x, nbr_fea, nbr_idx, Wl, bl, stats1, nullptr, nullptr, nullptr);
        finalize_bn<<<1, 128, 0, stream>>>(
            stats1, bn1_g + l * 128, bn1_b + l * 128,
            1.0f / (float)(NATOM * NBR_M), 128, s1, t1);
        conv_pass<true><<<2048, 256, 0, stream>>>(
            x, nbr_fea, nbr_idx, Wl, bl, nullptr, s1, t1, ns);
        hipMemsetAsync(stats2, 0, 128 * 4, stream);
        col_stats64<<<256, 256, 0, stream>>>(ns, NATOM, stats2);
        finalize_bn<<<1, 64, 0, stream>>>(
            stats2, bn2_g + l * 64, bn2_b + l * 64, 1.0f / (float)NATOM, 64, s2, t2);
        x_update<<<(NATOM * FEAT + 255) / 256, 256, 0, stream>>>(x, ns, s2, t2);
    }

    hipMemsetAsync(sums, 0, ((size_t)NCRYS * FEAT + NCRYS) * 4, stream);
    pool_kernel<<<(NATOM * FEAT + 255) / 256, 256, 0, stream>>>(x, cry_idx, sums, cnts);
    head_kernel<<<NCRYS, 256, 0, stream>>>(
        sums, cnts, fc_w, fc_b, head_w, head_b, out_w, out_b, out);
}
